// Conv2dKan_37958920962740
// MI455X (gfx1250) — compile-verified, matrix operand order checked by reading, not run on hardware
//
#include <hip/hip_runtime.h>
#include <hip/hip_bf16.h>

typedef __attribute__((ext_vector_type(16))) _Float16 v16h;
typedef __attribute__((ext_vector_type(8)))  _Float16 v8h_t;
typedef __attribute__((ext_vector_type(8)))  float    v8f;

#define CIN    16
#define COUT   32
#define HW     64
#define LTOT   4096      // 64*64 output positions
#define KK     9         // 3x3 taps
#define NFEAT  9         // 1 silu + 8 chebyshev
#define KDIM   1296      // CIN*KK*NFEAT
#define KPAD   1312      // padded to 41*32
#define NCHUNK 41

// ---------------------------------------------------------------------------
// Prep: fold w and w*c into W_eff[o][f] (f16), f = (i*9+k)*9 + n, zero-padded.
// ---------------------------------------------------------------------------
__global__ void kan_prep_weff(const float* __restrict__ w,
                              const float* __restrict__ c,
                              _Float16* __restrict__ weff) {
    int idx = blockIdx.x * blockDim.x + threadIdx.x;
    if (idx >= COUT * KPAD) return;
    int o = idx / KPAD;
    int f = idx - o * KPAD;
    float v = 0.0f;
    if (f < KDIM) {
        int n  = f % NFEAT;
        int ik = f / NFEAT;
        int i  = ik / KK;
        int k  = ik - i * KK;
        float wv = w[(i * COUT + o) * KK + k];             // w[i,o,k]
        if (n == 0) v = wv;
        else        v = wv * c[((i * COUT + o) * KK + k) * 8 + (n - 1)];
    }
    weff[o * KPAD + f] = (_Float16)v;
}

// ---------------------------------------------------------------------------
// Main: one (batch, 16-column) tile per 64-thread block (2 waves).
//   Phase 1: build F[col][KPAD] features in LDS (f16).
//   Phase 2: wave w computes cout rows 16w..16w+15 via 41 x wmma_f32_16x16x32_f16.
// ---------------------------------------------------------------------------
__global__ __launch_bounds__(64) void kan_main(const float* __restrict__ x,
                                               const _Float16* __restrict__ weff,
                                               float* __restrict__ out) {
    __shared__ _Float16 Flds[16 * KPAD];   // 41984 bytes

    const int blk   = blockIdx.x;
    const int b     = blk >> 8;            // 256 column-tiles per batch
    const int ltile = blk & 255;
    const int l0    = ltile * 16;
    const int tid   = threadIdx.x;

    // -------- Phase 1: feature generation into LDS --------
    // 144 (i,k) taps x 16 columns = 2304 patch values; 36 per thread.
    for (int pv = tid; pv < CIN * KK * 16; pv += 64) {
        int ik  = pv >> 4;
        int col = pv & 15;
        int i   = ik / KK;
        int k   = ik - i * KK;
        int l   = l0 + col;
        int oh  = l >> 6;
        int ow  = l & 63;
        int ih  = oh + (k / 3) - 1;
        int iw  = ow + (k % 3) - 1;
        float wx = 0.0f;
        if ((unsigned)ih < (unsigned)HW && (unsigned)iw < (unsigned)HW)
            wx = x[(((b * CIN + i) * HW) + ih) * HW + iw];

        // One exp serves both sigmoid and tanh.
        float wc  = fminf(fmaxf(wx, -20.0f), 20.0f);
        float e   = __expf(-wc);
        float sig = 1.0f / (1.0f + e);
        float silu = wx * sig;
        float e2  = e * e;
        float y   = (1.0f - e2) / (1.0f + e2);   // tanh(wx)

        _Float16* dst = &Flds[col * KPAD + ik * NFEAT];
        dst[0] = (_Float16)silu;
        // Chebyshev recurrence: T_n(y) = cos(n*arccos(y))
        float tm = 1.0f, tc = y;
        float y2 = 2.0f * y;
        dst[1] = (_Float16)tc;
#pragma unroll
        for (int n = 2; n <= 8; ++n) {
            float tn = y2 * tc - tm;
            tm = tc; tc = tn;
            dst[n] = (_Float16)tc;
        }
    }
    // Zero-pad feature rows KDIM..KPAD-1 (16 rows x 16 cols = 256 halves).
    {
        int z = tid * 4;
#pragma unroll
        for (int q = 0; q < 4; ++q) {
            int idx = z + q;                  // 0..255
            int col = idx >> 4;
            int f   = KDIM + (idx & 15);
            Flds[col * KPAD + f] = (_Float16)0.0f;
        }
    }
    __syncthreads();

    // -------- Phase 2: WMMA contraction --------
    const int wave = tid >> 5;        // 0,1 -> cout block
    const int lane = tid & 31;
    const int oblk = wave * 16;
    const int ncol = lane & 15;       // A: row m ; B: col n ; D: col n
    const int hi   = lane >> 4;
    const int s    = hi * 8;          // first K of halves 0..7 per 16-bit layout

    v8f acc = {};
    const _Float16* arow = weff + (size_t)(oblk + ncol) * KPAD;
    const _Float16* brow = &Flds[ncol * KPAD];

    for (int cch = 0; cch < NCHUNK; ++cch) {
        const int base = cch * 32 + s;
        v16h afrag, bfrag;
        // halves 0..7  <- K [base   .. base+7 ]
        // halves 8..15 <- K [base+16.. base+23]
        *(v8h_t*)&afrag       = *(const v8h_t*)(arow + base);
        *((v8h_t*)&afrag + 1) = *(const v8h_t*)(arow + base + 16);
        *(v8h_t*)&bfrag       = *(const v8h_t*)(brow + base);
        *((v8h_t*)&bfrag + 1) = *(const v8h_t*)(brow + base + 16);
        acc = __builtin_amdgcn_wmma_f32_16x16x32_f16(
            /*neg_a=*/false, afrag, /*neg_b=*/false, bfrag,
            /*c_mod=*/(short)0, acc, /*reuse_a=*/false, /*reuse_b=*/false);
    }

    // D layout: lane holds column n = lane&15; VGPR r holds row r + 8*hi.
    const int l = l0 + ncol;
#pragma unroll
    for (int r = 0; r < 8; ++r) {
        int m = oblk + r + hi * 8;
        out[((size_t)(b * COUT + m) * LTOT) + l] = acc[r];
    }
}

extern "C" void kernel_launch(void* const* d_in, const int* in_sizes, int n_in,
                              void* d_out, int out_size, void* d_ws, size_t ws_size,
                              hipStream_t stream) {
    const float* x = (const float*)d_in[0];   // (16,16,64,64)
    const float* w = (const float*)d_in[1];   // (16,32,9,1)
    const float* c = (const float*)d_in[2];   // (16,32,9,1,8)
    float* out = (float*)d_out;               // (16,32,64,64)

    _Float16* weff = (_Float16*)d_ws;         // 32*1312 halves = 83968 bytes

    {
        int total = COUT * KPAD;              // 41984
        int threads = 256;
        int blocks = (total + threads - 1) / threads;
        kan_prep_weff<<<blocks, threads, 0, stream>>>(w, c, weff);
    }
    {
        int blocks = 16 * 256;                // batch x column-tiles
        kan_main<<<blocks, 64, 0, stream>>>(x, weff, out);
    }
}
